// VolumeRenderer_47141561041601
// MI455X (gfx1250) — matrix operand order repaired and checked by reading, not compiled, not run on hardware
//
#include <hip/hip_runtime.h>

typedef __attribute__((ext_vector_type(16))) _Float16 v16h;
typedef __attribute__((ext_vector_type(8)))  float    v8f;

#define WPB 4            // waves (rays) per block
#define NS 64            // samples per ray
#define POS_IN 63
#define KPAD 64
#define HID 128
#define DIR_IN 27
#define COLOR_IN 155     // HID + DIR_IN
#define PW (POS_IN*HID)  // 8064
#define TOTAL_PARAMS 8789

__device__ __forceinline__ float fsigmoid(float x) { return 1.0f / (1.0f + __expf(-x)); }

// A fragment (16x32 f16, ISA 7.12.2): lanes 0-15 row M, K {0..7,16..23}; lanes 16-31 same M, K {8..15,24..31}
__device__ __forceinline__ v16h load_a_frag(const _Float16* row, int k0, int hi) {
  const _Float16* p = row + k0 + (hi ? 8 : 0);
  v16h a;
#pragma unroll
  for (int v = 0; v < 8; ++v) {
    int k = (v < 4) ? (2*v) : (16 + 2*(v-4));
    a[2*v]   = p[k];
    a[2*v+1] = p[k+1];
  }
  return a;
}

// B fragment (32x16 f16): lane = column N; lanes 0-15 K {0..7,16..23}, lanes 16-31 K {8..15,24..31}, 2 K per VGPR
__device__ __forceinline__ v16h load_b_frag(const _Float16* col, int k0, int hi) {
  const _Float16* p = col + k0 + (hi ? 8 : 0);
  v16h b;
#pragma unroll
  for (int v = 0; v < 8; ++v) {
    int k = (v < 4) ? (2*v) : (16 + 2*(v-4));
    b[2*v]   = p[k];
    b[2*v+1] = p[k+1];
  }
  return b;
}

__global__ void __launch_bounds__(32*WPB)
nerf_render_kernel(const float* __restrict__ params,
                   const float* __restrict__ poses,
                   const float* __restrict__ intr,
                   const int* __restrict__ Hp, const int* __restrict__ Wp,
                   float* __restrict__ out, int B)
{
  __shared__ _Float16 w_lds[HID*KPAD];          // pos_w as f16, K padded to 64
  __shared__ _Float16 pe_lds[WPB][16*KPAD];     // per-wave posenc tile
  __shared__ _Float16 h_lds[WPB][16*HID];       // per-wave hidden activations
  __shared__ float    comp[WPB][NS][4];         // sigma, r, g, b per sample
  __shared__ float    de_lds[WPB][32];          // direction encoding (27 used)
  __shared__ float    pos_b_s[HID];
  __shared__ float    sig_w_s[HID];
  __shared__ float    col_w_s[3*COLOR_IN];
  __shared__ float    cb_s[4];                  // col_b[0..2], sig_b

  const int H = Hp[0], W = Wp[0];
  const int HW = H * W;
  const int rays = B * HW;
  const int tid  = threadIdx.x;
  const int wave = tid >> 5;
  const int lane = tid & 31;

  const int rayRaw = (int)blockIdx.x * WPB + wave;
  int rayG = rayRaw < rays ? rayRaw : rays - 1;
  const int b   = rayG / HW;
  const int pix = rayG - b * HW;
  const int ii  = pix / W;
  const int jj  = pix - ii * W;

  // ---- stage per-batch parameters (block-uniform batch: HW % WPB == 0) ----
  int bblk = (int)blockIdx.x * WPB; if (bblk >= rays) bblk = rays - 1;
  bblk /= HW;
  const float* pp = params + (size_t)bblk * TOTAL_PARAMS;
  for (int idx = tid; idx < HID*KPAD; idx += blockDim.x) {
    int h = idx >> 6, k = idx & 63;
    w_lds[idx] = (k < POS_IN) ? (_Float16)pp[h*POS_IN + k] : (_Float16)0.0f;
  }
  if (tid < HID) { pos_b_s[tid] = pp[PW + tid]; sig_w_s[tid] = pp[PW + HID + tid]; }
  for (int idx = tid; idx < 3*COLOR_IN; idx += blockDim.x)
    col_w_s[idx] = pp[PW + 2*HID + 1 + idx];
  if (tid < 3)  cb_s[tid] = pp[PW + 2*HID + 1 + 3*COLOR_IN + tid]; // col_b
  if (tid == 3) cb_s[3]   = pp[PW + 2*HID];                        // sig_b
  __syncthreads();

  // ---- ray setup (all lanes of a wave share one ray) ----
  const float fx = intr[b*9+0], cx = intr[b*9+2], fy = intr[b*9+4], cy = intr[b*9+5];
  const float* P = poses + b*16;
  const float dcx = ((float)jj - cx) / fx;
  const float dcy = -((float)ii - cy) / fy;
  const float dcz = -1.0f;
  const float d0 = P[0]*dcx + P[1]*dcy + P[2]*dcz;
  const float d1 = P[4]*dcx + P[5]*dcy + P[6]*dcz;
  const float d2 = P[8]*dcx + P[9]*dcy + P[10]*dcz;
  const float o0 = P[3], o1 = P[7], o2 = P[11];

  // direction encoding: 27 features, one per lane
  if (lane < DIR_IN) {
    float dv[3] = {d0, d1, d2};
    float val;
    if (lane < 3) val = dv[lane];
    else {
      int k = (lane - 3) / 6, w = (lane - 3) % 6;
      float freq = __builtin_ldexpf(3.14159265358979f, k);
      float ang  = dv[(w < 3) ? w : (w - 3)] * freq;
      val = (w < 3) ? __sinf(ang) : __cosf(ang);
    }
    de_lds[wave][lane] = val;
  }

  const int sl = lane & 15;   // sample-in-tile / WMMA row M / col N
  const int g  = lane >> 4;   // lane-half
  const float dz = (10.0f - 0.1f) / (float)(NS - 1);

  for (int t = 0; t < 4; ++t) {
    // positional encoding for 16 samples: lanes 0-15 features 0..31, lanes 16-31 features 32..63
    {
      int s = t*16 + sl;
      float z = 0.1f + (float)s * dz;
      float pt[3] = { o0 + d0*z, o1 + d1*z, o2 + d2*z };
      _Float16* dst = &pe_lds[wave][sl * KPAD];
#pragma unroll
      for (int ff = 0; ff < 32; ++ff) {
        int f = g*32 + ff;
        float val;
        if (f < 3) val = pt[f];
        else if (f < POS_IN) {
          int k = (f - 3) / 6, w = (f - 3) % 6;
          float freq = __builtin_ldexpf(3.14159265358979f, k);
          float ang  = pt[(w < 3) ? w : (w - 3)] * freq;
          val = (w < 3) ? __sinf(ang) : __cosf(ang);
        } else val = 0.0f;
        dst[f] = (_Float16)val;
      }
    }
    __syncthreads();

    // hidden layer: h[16x128] = relu(pe[16x64] x Wt[64x128] + b) via v_wmma_f32_16x16x32_f16
    {
      const _Float16* prow = &pe_lds[wave][sl * KPAD];
      v16h a0 = load_a_frag(prow, 0, g);
      v16h a1 = load_a_frag(prow, 32, g);
#pragma unroll
      for (int n = 0; n < 8; ++n) {
        const _Float16* wr = &w_lds[(n*16 + sl) * KPAD];
        v16h b0 = load_b_frag(wr, 0, g);
        v16h b1 = load_b_frag(wr, 32, g);
        v8f c = {};
        c = __builtin_amdgcn_wmma_f32_16x16x32_f16(false, a0, false, b0, (short)0, c, false, false);
        c = __builtin_amdgcn_wmma_f32_16x16x32_f16(false, a1, false, b1, (short)0, c, false, false);
        int hh = n*16 + sl;   // D layout: N = lane%16
#pragma unroll
        for (int r = 0; r < 8; ++r) {
          int M = r + g*8;    // D layout: M = vgpr + 8*(lane>=16)
          float hv = c[r] + pos_b_s[hh];
          h_lds[wave][M*HID + hh] = (_Float16)fmaxf(hv, 0.0f);
        }
      }
    }
    __syncthreads();

    // heads: lanes 0-15 -> sigma + red; lanes 16-31 -> green + blue
    {
      const _Float16* hrow = &h_lds[wave][sl * HID];
      const float* dep = de_lds[wave];
      int s = t*16 + sl;
      if (g == 0) {
        float sg = cb_s[3];
        for (int c = 0; c < HID; ++c) sg += (float)hrow[c] * sig_w_s[c];
        float rr = cb_s[0];
        for (int c = 0; c < COLOR_IN; ++c) {
          float fv = (c < HID) ? (float)hrow[c] : dep[c - HID];
          rr += fv * col_w_s[c];
        }
        comp[wave][s][0] = sg;
        comp[wave][s][1] = fsigmoid(rr);
      } else {
        float gg = cb_s[1], bb = cb_s[2];
        for (int c = 0; c < COLOR_IN; ++c) {
          float fv = (c < HID) ? (float)hrow[c] : dep[c - HID];
          gg += fv * col_w_s[COLOR_IN + c];
          bb += fv * col_w_s[2*COLOR_IN + c];
        }
        comp[wave][s][2] = fsigmoid(gg);
        comp[wave][s][3] = fsigmoid(bb);
      }
    }
    __syncthreads();
  }

  // alpha-composite along the ray (serial scan, 64 steps)
  if (lane == 0 && rayRaw < rays) {
    float T = 1.0f, rc = 0.f, gc = 0.f, bc = 0.f, dsum = 0.f, asum = 0.f;
    for (int s = 0; s < NS; ++s) {
      float sg    = comp[wave][s][0];
      float dist  = (s == NS-1) ? 1e10f : dz;
      float alpha = 1.0f - __expf(-fmaxf(sg, 0.0f) * dist);
      float wgt   = alpha * T;
      rc   += wgt * comp[wave][s][1];
      gc   += wgt * comp[wave][s][2];
      bc   += wgt * comp[wave][s][3];
      dsum += wgt * (0.1f + (float)s * dz);
      asum += wgt;
      T *= (1.0f - alpha + 1e-10f);
    }
    float wb = 1.0f - asum;  // white background
    out[(size_t)(b*3 + 0)*HW + pix] = rc + wb;
    out[(size_t)(b*3 + 1)*HW + pix] = gc + wb;
    out[(size_t)(b*3 + 2)*HW + pix] = bc + wb;
    out[(size_t)B*3*HW + rayG] = dsum;
    out[(size_t)B*4*HW + rayG] = asum;
  }
}

extern "C" void kernel_launch(void* const* d_in, const int* in_sizes, int n_in,
                              void* d_out, int out_size, void* d_ws, size_t ws_size,
                              hipStream_t stream) {
  (void)n_in; (void)d_ws; (void)ws_size;
  const float* params = (const float*)d_in[0];
  const float* poses  = (const float*)d_in[1];
  const float* intr   = (const float*)d_in[2];
  const int*   Hp     = (const int*)d_in[3];
  const int*   Wp     = (const int*)d_in[4];
  float* out = (float*)d_out;

  int B  = in_sizes[1] / 16;          // camera_poses: B x 4 x 4
  int HW = out_size / (5 * B);        // out = (3+1+1) * B * H * W
  int rays = B * HW;
  int blocks = (rays + WPB - 1) / WPB;

  nerf_render_kernel<<<blocks, 32*WPB, 0, stream>>>(params, poses, intr, Hp, Wp, out, B);
}
